// AttentiveSememe_78151224918146
// MI455X (gfx1250) — matrix operand order repaired and checked by reading, not compiled, not run on hardware
//
#include <hip/hip_runtime.h>
#include <hip/hip_bf16.h>
#include <math.h>

typedef _Float16 half_t;
typedef __attribute__((ext_vector_type(16))) _Float16 v16h;
typedef __attribute__((ext_vector_type(8)))  _Float16 v8h;
typedef __attribute__((ext_vector_type(8)))  float    v8f;

// Problem constants
#define BB   1024
#define LL   18
#define EE   300
#define HH   300
#define KX   640           // 2*E (600) padded to 64-mult
#define KH   320           // H (300) padded to 32-mult
#define NP_H 320           // H (300) padded to 64-mult
#define GS   320           // per-gate padded width
#define GP   1280          // 4 gates * GS (padded 4H layout)
#define NT   4             // 64-wide N strip per wave
#define MT   2             // 32-tall M strip per wave (B-fragment reuse x2)

union AFrag { v16h v; v8h h[2]; };

// ---------------------------------------------------------------------------
// Generic WMMA GEMM: D[M,N] = A[M,Kp](f16) * W[Np,Kp]^T(f16) (+bias) (relu?)
// one wave per block, 32x64 output tile, f32 accumulation.
// A-fragments reused across 4 N-tiles; B-fragments reused across 2 M-tiles.
// ---------------------------------------------------------------------------
__global__ void __launch_bounds__(32)
wmma_gemm(const half_t* __restrict__ A, int lda,
          const half_t* __restrict__ W, int ldw,
          const float*  __restrict__ bias1,
          float* __restrict__ D, long ldd,
          int N, int Kp, int relu)
{
    const int lane = threadIdx.x;
    const int lcol = lane & 15;
    const int lhi  = lane >> 4;
    const int row0 = blockIdx.y * (16 * MT);
    const int col0 = blockIdx.x * (16 * NT);

    v8f acc[MT][NT];
    #pragma unroll
    for (int mt = 0; mt < MT; ++mt)
        #pragma unroll
        for (int nt = 0; nt < NT; ++nt) { v8f z = {}; acc[mt][nt] = z; }

    // A fragment: lanes 0-15 -> K sub-block 0; lanes 16-31 -> K sub-block +8
    const half_t* arow0 = A + (long)(row0 + lcol) * lda + lhi * 8;
    const half_t* arow1 = arow0 + (long)16 * lda;

    for (int kb = 0; kb < Kp; kb += 32) {
        AFrag a0, a1;
        a0.h[0] = *(const v8h*)(arow0 + kb);
        a0.h[1] = *(const v8h*)(arow0 + kb + 16);
        a1.h[0] = *(const v8h*)(arow1 + kb);
        a1.h[1] = *(const v8h*)(arow1 + kb + 16);
        #pragma unroll
        for (int nt = 0; nt < NT; ++nt) {
            const half_t* brow = W + (long)(col0 + nt * 16 + lcol) * ldw + kb + lhi * 8;
            AFrag b;
            b.h[0] = *(const v8h*)(brow);
            b.h[1] = *(const v8h*)(brow + 16);
            acc[0][nt] = __builtin_amdgcn_wmma_f32_16x16x32_f16(
                false, a0.v, false, b.v, (short)0, acc[0][nt], false, false);
            acc[1][nt] = __builtin_amdgcn_wmma_f32_16x16x32_f16(
                false, a1.v, false, b.v, (short)0, acc[1][nt], false, false);
        }
    }

    #pragma unroll
    for (int nt = 0; nt < NT; ++nt) {
        const int col = col0 + nt * 16 + lcol;
        if (col < N) {
            const float bv = bias1 ? bias1[col] : 0.f;
            #pragma unroll
            for (int mt = 0; mt < MT; ++mt) {
                #pragma unroll
                for (int r = 0; r < 8; ++r) {
                    const int m = row0 + mt * 16 + r + 8 * lhi;
                    float v = acc[mt][nt][r] + bv;
                    if (relu) v = v > 0.f ? v : 0.f;
                    D[(long)m * ldd + col] = v;
                }
            }
        }
    }
}

// ---------------------------------------------------------------------------
// Fused recurrent step: for a 16x16 (batch, j) tile, the wave's 4 WMMA tiles
// are the 4 gates (per-gate padded layout, columns g*GS + j0).
//   acc[g] = xg[:, t, g*GS+j0 .. +15]  (C-operand)  +  h @ WhhP^T
// then sigmoid/tanh, cell update, h emitted as f16 (state + Vh slot).
// ---------------------------------------------------------------------------
__global__ void __launch_bounds__(32)
wmma_lstm_step(const half_t* __restrict__ h16, const half_t* __restrict__ WhhP,
               const float* __restrict__ xgt, long ldxg,   // xg slice at time t
               float* __restrict__ cst,
               half_t* __restrict__ h16out,
               half_t* __restrict__ vh,
               int t, int dofs, int first)
{
    const int lane = threadIdx.x;
    const int lcol = lane & 15;
    const int lhi  = lane >> 4;
    const int row0 = blockIdx.y * 16;   // batch tile
    const int j0   = blockIdx.x * 16;   // hidden-unit tile (0..288)

    v8f acc[4];
    #pragma unroll
    for (int g = 0; g < 4; ++g) {
        const int col = g * GS + j0 + lcol;
        v8f z;
        #pragma unroll
        for (int r = 0; r < 8; ++r)
            z[r] = xgt[(long)(row0 + r + 8 * lhi) * ldxg + col];
        acc[g] = z;
    }

    if (!first) {  // uniform branch; EXEC stays all-ones for WMMA
        const half_t* arow = h16 + (long)(row0 + lcol) * KH + lhi * 8;
        for (int kb = 0; kb < KH; kb += 32) {
            AFrag a;
            a.h[0] = *(const v8h*)(arow + kb);
            a.h[1] = *(const v8h*)(arow + kb + 16);
            #pragma unroll
            for (int g = 0; g < 4; ++g) {
                const half_t* brow = WhhP + (long)(g * GS + j0 + lcol) * KH + kb + lhi * 8;
                AFrag b;
                b.h[0] = *(const v8h*)(brow);
                b.h[1] = *(const v8h*)(brow + 16);
                acc[g] = __builtin_amdgcn_wmma_f32_16x16x32_f16(
                    false, a.v, false, b.v, (short)0, acc[g], false, false);
            }
        }
    }

    const int j = j0 + lcol;
    if (j < HH) {
        #pragma unroll
        for (int r = 0; r < 8; ++r) {
            const int m = row0 + r + 8 * lhi;
            const float iv = 1.f / (1.f + expf(-acc[0][r]));
            const float fv = 1.f / (1.f + expf(-acc[1][r]));
            const float gv = tanhf(acc[2][r]);
            const float ov = 1.f / (1.f + expf(-acc[3][r]));
            const float cold = first ? 0.f : cst[(long)m * HH + j];
            const float cn = fv * cold + iv * gv;
            cst[(long)m * HH + j] = cn;
            const float hv = ov * tanhf(cn);
            h16out[(long)m * KH + j] = (half_t)hv;
            vh[((long)m * LL + t) * KX + dofs + j] = (half_t)hv;
        }
    }
}

// ---------------------------------------------------------------------------
// Gather + concat: x[b,l] = [ (sem==0 ? s : w) , s , zero-pad ] -> f16 (KX wide)
// ---------------------------------------------------------------------------
__global__ void embed_kernel(const int* __restrict__ word, const int* __restrict__ sem,
                             const float* __restrict__ wt, const float* __restrict__ st,
                             half_t* __restrict__ xh)
{
    const int bl = blockIdx.x;              // 0..B*L-1
    const int b  = bl / LL;
    const int l  = bl - b * LL;
    const int sidx = sem[b * LL + l];
    const int widx = word[b];
    const int col = threadIdx.x;            // 0..639
    float v = 0.f;
    if (col < EE) {
        const float sv = st[(long)sidx * EE + col];
        const float wv = wt[(long)widx * EE + col];
        v = (sidx == 0) ? sv : wv;
    } else if (col < 2 * EE) {
        v = st[(long)sidx * EE + (col - EE)];
    }
    xh[(long)bl * KX + col] = (half_t)v;
}

// a_g = mean_l(sem_table[sememes]) -> f16 padded to KH
__global__ void ag_kernel(const int* __restrict__ sem, const float* __restrict__ st,
                          half_t* __restrict__ ag)
{
    const int b = blockIdx.x;
    const int e = threadIdx.x;              // 0..319
    float s = 0.f;
    if (e < EE) {
        for (int l = 0; l < LL; ++l)
            s += st[(long)sem[b * LL + l] * EE + e];
        s *= (1.f / LL);
    }
    ag[(long)b * KH + e] = (half_t)s;
}

// f32 -> f16 with zero padding: dst[rows_p, cols_p] from src[rows, cols]
__global__ void cvt_pad(const float* __restrict__ src, half_t* __restrict__ dst,
                        int rows, int cols, int rows_p, int cols_p)
{
    const long n = (long)rows_p * cols_p;
    for (long i = blockIdx.x * (long)blockDim.x + threadIdx.x; i < n;
         i += (long)gridDim.x * blockDim.x) {
        const int r  = (int)(i / cols_p);
        const int cc = (int)(i - (long)r * cols_p);
        const float v = (r < rows && cc < cols) ? src[(long)r * cols + cc] : 0.f;
        dst[i] = (half_t)v;
    }
}

// f32 -> f16, rows re-laid per-gate: dst[4*GS, cols_p] from src[4*HH, cols]
__global__ void cvt_pad_gates(const float* __restrict__ src, half_t* __restrict__ dst,
                              int cols, int cols_p)
{
    const long n = (long)GP * cols_p;
    for (long i = blockIdx.x * (long)blockDim.x + threadIdx.x; i < n;
         i += (long)gridDim.x * blockDim.x) {
        const int p  = (int)(i / cols_p);      // padded row
        const int cc = (int)(i - (long)p * cols_p);
        const int g  = p / GS;
        const int j  = p - g * GS;
        float v = 0.f;
        if (j < HH && cc < cols) v = src[(long)(g * HH + j) * cols + cc];
        dst[i] = (half_t)v;
    }
}

// padded combined bias: biasP[g*GS+j] = bih[g*HH+j] + bhh[g*HH+j]
__global__ void bias_pad(const float* __restrict__ bih, const float* __restrict__ bhh,
                         float* __restrict__ biasP)
{
    const int p = blockIdx.x * blockDim.x + threadIdx.x;
    if (p >= GP) return;
    const int g = p / GS;
    const int j = p - g * GS;
    biasP[p] = (j < HH) ? (bih[g * HH + j] + bhh[g * HH + j]) : 0.f;
}

__global__ void zero_u32(unsigned int* __restrict__ p, long n)
{
    for (long i = blockIdx.x * (long)blockDim.x + threadIdx.x; i < n;
         i += (long)gridDim.x * blockDim.x)
        p[i] = 0u;
}

// ---------------------------------------------------------------------------
extern "C" void kernel_launch(void* const* d_in, const int* in_sizes, int n_in,
                              void* d_out, int out_size, void* d_ws, size_t ws_size,
                              hipStream_t stream)
{
    const int*   word  = (const int*)  d_in[0];
    const int*   sem   = (const int*)  d_in[1];
    const float* wtab  = (const float*)d_in[2];
    const float* stab  = (const float*)d_in[3];
    const float* Wih[2] = { (const float*)d_in[4], (const float*)d_in[8] };
    const float* Whh[2] = { (const float*)d_in[5], (const float*)d_in[9] };
    const float* bih[2] = { (const float*)d_in[6], (const float*)d_in[10] };
    const float* bhh[2] = { (const float*)d_in[7], (const float*)d_in[11] };
    const float* Wa = (const float*)d_in[12];
    const float* ba = (const float*)d_in[13];
    const float* Wb = (const float*)d_in[14];
    const float* bb = (const float*)d_in[15];
    float* out  = (float*)d_out;            // V (B,L,H) then v_g (B,E)
    float* outV = out;
    float* outG = out + (long)BB * LL * HH;

    // Workspace carve-up (256B aligned)
    char* ws = (char*)d_ws;
    size_t o = 0;
    auto carve = [&](size_t bytes) -> char* {
        char* p = ws + o;
        o = (o + bytes + 255) & ~(size_t)255;
        return p;
    };
    half_t* xh    = (half_t*)carve((size_t)BB * LL * KX * 2);   // 23.6 MB
    half_t* WihP  = (half_t*)carve((size_t)GP * KX * 2);        // 1.6 MB (per dir)
    half_t* WhhP  = (half_t*)carve((size_t)GP * KH * 2);        // 0.8 MB (per dir)
    half_t* Wa16  = (half_t*)carve((size_t)NP_H * KX * 2);
    half_t* Wb16  = (half_t*)carve((size_t)NP_H * KH * 2);
    float*  biasP = (float*) carve((size_t)GP * 4);
    float*  xg    = (float*) carve((size_t)BB * LL * GP * 4);   // 94.4 MB (per dir)
    half_t* h16   = (half_t*)carve((size_t)BB * KH * 2);
    float*  cst   = (float*) carve((size_t)BB * HH * 4);
    half_t* vh    = (half_t*)carve((size_t)BB * LL * KX * 2);   // 23.6 MB
    half_t* ag16  = (half_t*)carve((size_t)BB * KH * 2);
    (void)ws_size; (void)in_sizes; (void)n_in; (void)out_size;

    const dim3 blk32(32);

    // 1) embeddings / concat input, global-average branch
    embed_kernel<<<BB * LL, KX, 0, stream>>>(word, sem, wtab, stab, xh);
    ag_kernel<<<BB, KH, 0, stream>>>(sem, stab, ag16);

    // 2) small weight conversions
    cvt_pad<<<512, 256, 0, stream>>>(Wa, Wa16, HH, 2 * EE, NP_H, KX);
    cvt_pad<<<512, 256, 0, stream>>>(Wb, Wb16, EE, EE, NP_H, KH);

    // 3) v_g = relu(a_g @ Wb^T + bb)   (M=1024, N=300, K=320)
    {
        dim3 grid((HH + 16 * NT - 1) / (16 * NT), BB / (16 * MT));
        wmma_gemm<<<grid, blk32, 0, stream>>>(ag16, KH, Wb16, KH, bb,
                                              outG, HH, HH, KH, 1);
    }

    // 4) zero Vh (pad cols must be 0 for the final GEMM)
    zero_u32<<<2048, 256, 0, stream>>>((unsigned int*)vh, (long)BB * LL * KX * 2 / 4);

    // 5) both LSTM directions
    for (int dir = 0; dir < 2; ++dir) {
        cvt_pad_gates<<<2048, 256, 0, stream>>>(Wih[dir], WihP, 2 * EE, KX);
        cvt_pad_gates<<<1024, 256, 0, stream>>>(Whh[dir], WhhP, HH, KH);
        bias_pad<<<(GP + 255) / 256, 256, 0, stream>>>(bih[dir], bhh[dir], biasP);

        // xg = x @ Wih^T + (bih+bhh), per-gate padded cols (M=18432, N=1280, K=640)
        {
            dim3 grid(GP / (16 * NT), (BB * LL) / (16 * MT));
            wmma_gemm<<<grid, blk32, 0, stream>>>(xh, KX, WihP, KX, biasP,
                                                  xg, GP, GP, KX, 0);
        }

        // keep h16 pad columns deterministic/zero for recurrent A loads
        zero_u32<<<512, 256, 0, stream>>>((unsigned int*)h16, (long)BB * KH * 2 / 4);

        for (int step = 0; step < LL; ++step) {
            const int t = dir ? (LL - 1 - step) : step;
            dim3 grid((HH + 15) / 16, BB / 16);   // 19 x 64
            wmma_lstm_step<<<grid, blk32, 0, stream>>>(
                h16, WhhP, xg + (long)t * GP, (long)LL * GP,
                cst, h16, vh, t, dir * HH, step == 0 ? 1 : 0);
        }
    }

    // 6) V = relu(Vh @ Wa^T + ba)   (M=18432, N=300, K=640)
    {
        dim3 grid((HH + 16 * NT - 1) / (16 * NT), (BB * LL) / (16 * MT));
        wmma_gemm<<<grid, blk32, 0, stream>>>(vh, KX, Wa16, KX, ba,
                                              outV, HH, HH, KX, 1);
    }
}
